// tranformer_layer_54331336294407
// MI455X (gfx1250) — matrix operand order repaired and checked by reading, not compiled, not run on hardware
//
#include <hip/hip_runtime.h>
#include <math.h>

// ---------------------------------------------------------------------------
// Transformer layer for MI455X (gfx1250, wave32, WMMA bf16 16x16x32).
// All GEMMs bf16 WMMA w/ fp32 accum; LN/softmax/GELU fp32.
// Weights pre-transposed to [N][K] so LDS staging of A and B tiles is a pure
// contiguous copy -> GLOBAL_LOAD_ASYNC_TO_LDS_B128 (ASYNCcnt) + double buffer.
// ---------------------------------------------------------------------------

typedef __attribute__((ext_vector_type(16))) __bf16 v16bf;
typedef __attribute__((ext_vector_type(8)))  float  v8f;
typedef int v4i __attribute__((vector_size(16)));   // matches builtin param type

#define WMMA_BF16(a, b, c) \
    __builtin_amdgcn_wmma_f32_16x16x32_bf16(false, (a), false, (b), (short)0, (c), false, false)

// ------------------------- async copy to LDS (guarded) ----------------------
#if defined(__AMDGCN__) && __has_builtin(__builtin_amdgcn_global_load_async_to_lds_b128)
#define HAS_ASYNC_LDS 1
#else
#define HAS_ASYNC_LDS 0
#endif

__device__ __forceinline__ void cp16_to_lds(void* lds, const void* g) {
#if HAS_ASYNC_LDS
    typedef v4i __attribute__((address_space(1))) gas_v4i;
    typedef v4i __attribute__((address_space(3))) las_v4i;
    __builtin_amdgcn_global_load_async_to_lds_b128(
        (gas_v4i*)(void*)g, (las_v4i*)lds, 0, 0);
#else
    *(uint4*)lds = *(const uint4*)g;
#endif
}

__device__ __forceinline__ void async_wait0() {
#if HAS_ASYNC_LDS
#if __has_builtin(__builtin_amdgcn_s_wait_asynccnt)
    __builtin_amdgcn_s_wait_asynccnt(0);
#else
    asm volatile("s_wait_asynccnt 0x0" ::: "memory");
#endif
#endif
}

// ------------------------- fragment gathers (from LDS) ----------------------
// A-matrix 16x32 bf16 (ISA 7.12.2): lane = g*16+m, VGPR j holds
// K = ((j>>2)<<4) + g*8 + 2*(j&3) (+1)  ==> two contiguous 16B runs per lane.
__device__ __forceinline__ v16bf load_frag_a(const __bf16* s, int rowBase, int stride) {
    const int lane = threadIdx.x & 31;
    const int g = lane >> 4, m = lane & 15;
    union { v16bf v; uint4 q[2]; } f;
    const __bf16* p = s + (rowBase + m) * stride + g * 8;
    f.q[0] = *(const uint4*)p;         // K = g*8 + 0..7
    f.q[1] = *(const uint4*)(p + 16);  // K = 16 + g*8 + 0..7
    return f.v;
}

// B-matrix 32x16 bf16: lane = g*16+n, same K order as A:
// K = g*8 + 0..7 and 16 + g*8 + 0..7.  Source staged [col][K].
__device__ __forceinline__ v16bf load_frag_b(const __bf16* s, int colBase, int stride) {
    const int lane = threadIdx.x & 31;
    const int g = lane >> 4, n = lane & 15;
    union { v16bf v; uint4 q[2]; } f;
    const __bf16* p = s + (colBase + n) * stride + g * 8;
    f.q[0] = *(const uint4*)p;
    f.q[1] = *(const uint4*)(p + 16);
    return f.v;
}

__device__ __forceinline__ float gelu_tanh(float x) {
    const float c = 0.7978845608028654f; // sqrt(2/pi)
    return 0.5f * x * (1.0f + tanhf(c * (x + 0.044715f * x * x * x)));
}

// ------------------ weight convert + transpose: [K,N]f32 -> [N,K]bf16 -------
__global__ __launch_bounds__(256) void cvt_transpose_kernel(
    const float* __restrict__ in, __bf16* __restrict__ out, int K, int N) {
    __shared__ __bf16 tile[32][33];
    const int k0 = blockIdx.y * 32, n0 = blockIdx.x * 32;
    const int tx = threadIdx.x & 31, ty = threadIdx.x >> 5;
#pragma unroll
    for (int i = 0; i < 4; ++i)
        tile[ty + i * 8][tx] = (__bf16)in[(size_t)(k0 + ty + i * 8) * N + n0 + tx];
    __syncthreads();
#pragma unroll
    for (int i = 0; i < 4; ++i)
        out[(size_t)(n0 + ty + i * 8) * K + k0 + tx] = tile[tx][ty + i * 8];
}

// ------------------------------- LayerNorm ----------------------------------
__global__ __launch_bounds__(256) void ln_bf16_kernel(
    const float* __restrict__ x, __bf16* __restrict__ out, int D) {
    const int row = blockIdx.x;
    const float* xr = x + (size_t)row * D;
    float s = 0.f, s2 = 0.f;
    for (int i = threadIdx.x; i < D; i += 256) {
        float v = xr[i];
        s += v; s2 += v * v;
    }
#pragma unroll
    for (int m = 16; m >= 1; m >>= 1) {
        s  += __shfl_xor(s, m, 32);
        s2 += __shfl_xor(s2, m, 32);
    }
    __shared__ float red[2][8];
    const int w = threadIdx.x >> 5;
    if ((threadIdx.x & 31) == 0) { red[0][w] = s; red[1][w] = s2; }
    __syncthreads();
    float ts = 0.f, ts2 = 0.f;
#pragma unroll
    for (int i = 0; i < 8; ++i) { ts += red[0][i]; ts2 += red[1][i]; }
    const float mu  = ts / (float)D;
    const float var = ts2 / (float)D - mu * mu;
    const float rinv = rsqrtf(var + 1e-5f);
    __bf16* o = out + (size_t)row * D;
    for (int i = threadIdx.x; i < D; i += 256)
        o[i] = (__bf16)((xr[i] - mu) * rinv);
}

// --------------------------------- GEMM -------------------------------------
// C[M,N] = epi(A[M,K] x Bt[N,K]^T) ; 128x128 block tile, 8 waves (64x32 each),
// K-step 32, double-buffered async LDS staging.
#define GS 40  // LDS row stride (elements): 80B rows, 16B aligned

template <int BIAS, int ACT, int RESID, int OUTF32>
__global__ __launch_bounds__(256) void gemm_bf16_kernel(
    const __bf16* __restrict__ A, const __bf16* __restrict__ Bt,
    int N, int K,
    const float* __restrict__ bias, const float* __restrict__ resid,
    float* __restrict__ outF, __bf16* __restrict__ outB) {
    __shared__ __bf16 As[2][128 * GS];
    __shared__ __bf16 Bs[2][128 * GS];

    const int t = threadIdx.x;
    const int m0 = blockIdx.y * 128;
    const int n0 = blockIdx.x * 128;
    const int wave = t >> 5;
    const int wm = wave & 1;   // 2 wave rows -> 64 rows each
    const int wn = wave >> 1;  // 4 wave cols -> 32 cols each

    v8f acc[4][2] = {};

    // stage one 128x32 tile of A and of Bt (each: 128 rows x 64B, contiguous)
    auto stage = [&](int buf, int kt) {
#pragma unroll
        for (int i = 0; i < 2; ++i) {
            const int idx = t + i * 256;           // 0..511
            const int r = idx >> 2, s4 = idx & 3;  // row, 16B segment
            const int lo = r * GS + s4 * 8;
            cp16_to_lds(&As[buf][lo], A  + (size_t)(m0 + r) * K + kt + s4 * 8);
            cp16_to_lds(&Bs[buf][lo], Bt + (size_t)(n0 + r) * K + kt + s4 * 8);
        }
    };

    stage(0, 0);
    async_wait0();
    __syncthreads();

    int buf = 0;
    for (int kt = 0; kt < K; kt += 32) {
        if (kt + 32 < K) stage(buf ^ 1, kt + 32);

        v16bf af[4], bf[2];
#pragma unroll
        for (int ms = 0; ms < 4; ++ms) af[ms] = load_frag_a(As[buf], wm * 64 + ms * 16, GS);
#pragma unroll
        for (int ns = 0; ns < 2; ++ns) bf[ns] = load_frag_b(Bs[buf], wn * 32 + ns * 16, GS);
#pragma unroll
        for (int ms = 0; ms < 4; ++ms)
#pragma unroll
            for (int ns = 0; ns < 2; ++ns)
                acc[ms][ns] = WMMA_BF16(af[ms], bf[ns], acc[ms][ns]);

        async_wait0();
        __syncthreads();
        buf ^= 1;
    }

    // epilogue (compile-time specialized): lane (g,n) holds rows 8g+r, col n
    const int lane = t & 31;
    const int g = lane >> 4, n = lane & 15;
#pragma unroll
    for (int ms = 0; ms < 4; ++ms)
#pragma unroll
        for (int ns = 0; ns < 2; ++ns) {
            const int row0 = m0 + wm * 64 + ms * 16 + g * 8;
            const int col  = n0 + wn * 32 + ns * 16 + n;
            float bv = 0.f;
            if (BIAS) bv = bias[col];
#pragma unroll
            for (int r = 0; r < 8; ++r) {
                const size_t off = (size_t)(row0 + r) * N + col;
                float v = acc[ms][ns][r];
                if (BIAS)  v += bv;
                if (ACT)   v = gelu_tanh(v);
                if (RESID) v += resid[off];
                if (OUTF32) outF[off] = v;
                else        outB[off] = (__bf16)v;
            }
        }
}

// ------------------------------ Attention -----------------------------------
// Flash-style: block = 128 queries of one (b,h); 8 waves x 16 query rows, so
// softmax stats stay intra-wave (16-lane shfl reductions match C layout).
#define SLEN 2048
#define LDQKV 3072
#define QS 80  // padded LDS stride: 160B rows, 16B aligned

__global__ __launch_bounds__(256) void attn_kernel(
    const __bf16* __restrict__ qkv, __bf16* __restrict__ out) {
    const int qb = blockIdx.x;
    const int h  = blockIdx.y;
    const int b  = blockIdx.z;
    const size_t base = (size_t)b * SLEN * LDQKV;
    const int hc = h * 64;

    const int wave = threadIdx.x >> 5;
    const int lane = threadIdx.x & 31;
    const int g = lane >> 4, n16 = lane & 15;

    __shared__ __bf16 Qs[128 * QS];
    __shared__ __bf16 Ks[64 * QS];
    __shared__ __bf16 Vt[64 * QS];       // transposed: [dim][key]
    __shared__ __bf16 Ps[8][16 * QS];    // per-wave P scratch

    // stage Q tile (128 rows x 128B, contiguous) via async copies
#pragma unroll
    for (int i = 0; i < 4; ++i) {
        const int idx = threadIdx.x + i * 256;  // 0..1023
        const int r = idx >> 3, s8 = idx & 7;
        cp16_to_lds(&Qs[r * QS + s8 * 8],
                    qkv + base + (size_t)(qb * 128 + r) * LDQKV + hc + s8 * 8);
    }
    async_wait0();
    __syncthreads();

    v16bf qf[2];
    qf[0] = load_frag_a(Qs,      wave * 16, QS);
    qf[1] = load_frag_a(Qs + 32, wave * 16, QS);

    float m_i[8], l_i[8];
    v8f o[4] = {};
#pragma unroll
    for (int r = 0; r < 8; ++r) { m_i[r] = -1e30f; l_i[r] = 0.f; }

    for (int kb = 0; kb < SLEN / 64; ++kb) {
        // K tile: natural [key][dim] == Bt layout for Q*K^T -> async copy
#pragma unroll
        for (int i = 0; i < 2; ++i) {
            const int idx = threadIdx.x + i * 256;  // 0..511
            const int r = idx >> 3, s8 = idx & 7;
            cp16_to_lds(&Ks[r * QS + s8 * 8],
                        qkv + base + (size_t)(kb * 64 + r) * LDQKV + 1024 + hc + s8 * 8);
        }
        // V tile: needs transpose -> vectorized load + b16 scatter
#pragma unroll
        for (int i = 0; i < 2; ++i) {
            const int idx = threadIdx.x + i * 256;
            const int key = idx >> 3, dg = idx & 7;
            union { uint4 q; __bf16 e[8]; } v;
            v.q = *(const uint4*)(qkv + base + (size_t)(kb * 64 + key) * LDQKV + 2048 + hc + dg * 8);
#pragma unroll
            for (int j = 0; j < 8; ++j)
                Vt[(dg * 8 + j) * QS + key] = v.e[j];
        }
        async_wait0();
        __syncthreads();

        // S = Q x K^T  (16 x 64 per wave)
        v8f sacc[4] = {};
#pragma unroll
        for (int ns = 0; ns < 4; ++ns)
#pragma unroll
            for (int c = 0; c < 2; ++c) {
                v16bf kf = load_frag_b(Ks + 32 * c, ns * 16, QS);
                sacc[ns] = WMMA_BF16(qf[c], kf, sacc[ns]);
            }

        // online softmax (row M = 8g + r lives in VGPR r of lane group g)
        const float scale = 0.125f;  // 1/sqrt(64)
        float pl[4][8];
#pragma unroll
        for (int r = 0; r < 8; ++r) {
            float mx = -1e30f;
#pragma unroll
            for (int ns = 0; ns < 4; ++ns) {
                const float v = sacc[ns][r] * scale;
                pl[ns][r] = v;
                mx = fmaxf(mx, v);
            }
#pragma unroll
            for (int msk = 8; msk >= 1; msk >>= 1)
                mx = fmaxf(mx, __shfl_xor(mx, msk, 32));
            const float mnew = fmaxf(m_i[r], mx);
            const float corr = __expf(m_i[r] - mnew);
            float rs = 0.f;
#pragma unroll
            for (int ns = 0; ns < 4; ++ns) {
                const float p = __expf(pl[ns][r] - mnew);
                pl[ns][r] = p;
                rs += p;
            }
#pragma unroll
            for (int msk = 8; msk >= 1; msk >>= 1)
                rs += __shfl_xor(rs, msk, 32);
            l_i[r] = l_i[r] * corr + rs;
            m_i[r] = mnew;
#pragma unroll
            for (int d = 0; d < 4; ++d) o[d][r] *= corr;
        }

        // P: C-layout regs -> A-layout via per-wave LDS scratch (in-order DS)
#pragma unroll
        for (int ns = 0; ns < 4; ++ns)
#pragma unroll
            for (int r = 0; r < 8; ++r)
                Ps[wave][(g * 8 + r) * QS + ns * 16 + n16] = (__bf16)pl[ns][r];

        v16bf pf[2];
        pf[0] = load_frag_a(&Ps[wave][0],      0, QS);
        pf[1] = load_frag_a(&Ps[wave][0] + 32, 0, QS);

        // O += P x V
#pragma unroll
        for (int d = 0; d < 4; ++d)
#pragma unroll
            for (int c = 0; c < 2; ++c) {
                v16bf vf = load_frag_b(Vt + 32 * c, d * 16, QS);
                o[d] = WMMA_BF16(pf[c], vf, o[d]);
            }
        __syncthreads();  // Ks/Vt restaged next iteration
    }

    // normalize and write bf16 [token][h*64 + dim]
#pragma unroll
    for (int d = 0; d < 4; ++d) {
#pragma unroll
        for (int r = 0; r < 8; ++r) {
            const int row = qb * 128 + wave * 16 + g * 8 + r;
            const float val = o[d][r] / l_i[r];
            out[((size_t)b * SLEN + row) * 1024 + hc + d * 16 + n16] = (__bf16)val;
        }
    }
}

// ------------------------------ launcher ------------------------------------
extern "C" void kernel_launch(void* const* d_in, const int* in_sizes, int n_in,
                              void* d_out, int out_size, void* d_ws, size_t ws_size,
                              hipStream_t stream) {
    (void)in_sizes; (void)n_in; (void)out_size; (void)ws_size;
    const float* x     = (const float*)d_in[0];
    const float* w_qkv = (const float*)d_in[1];
    const float* w_fc  = (const float*)d_in[2];
    const float* w1    = (const float*)d_in[3];
    const float* b1    = (const float*)d_in[4];
    const float* w2    = (const float*)d_in[5];
    const float* b2    = (const float*)d_in[6];
    float* out = (float*)d_out;

    char* ws = (char*)d_ws;
    const size_t MB = 1024ull * 1024ull;
    __bf16* hln    = (__bf16*)(ws +  0 * MB);  // 8 MB  [4096,1024] (reused for LN2)
    __bf16* wqkvT  = (__bf16*)(ws +  8 * MB);  // 6 MB  [3072,1024]
    __bf16* qkv    = (__bf16*)(ws + 14 * MB);  // 24 MB [4096,3072]
    __bf16* attn_o = (__bf16*)(ws + 38 * MB);  // 8 MB  [4096,1024]
    __bf16* wfcT   = (__bf16*)(ws + 46 * MB);  // 2 MB  [1024,1024]
    float*  x1     = (float*)(ws + 48 * MB);   // 16 MB [4096,1024]
    __bf16* w1T    = (__bf16*)(ws + 64 * MB);  // 8 MB  [4096,1024]
    __bf16* w2T    = (__bf16*)(ws + 72 * MB);  // 8 MB  [1024,4096]
    __bf16* mlph   = (__bf16*)(ws + 80 * MB);  // 32 MB [4096,4096]

    const int R = 2 * 2048;  // 4096 token rows

    // weights: fp32 [K,N] -> bf16 [N,K]
    cvt_transpose_kernel<<<dim3(3072 / 32, 1024 / 32), 256, 0, stream>>>(w_qkv, wqkvT, 1024, 3072);
    cvt_transpose_kernel<<<dim3(1024 / 32, 1024 / 32), 256, 0, stream>>>(w_fc,  wfcT,  1024, 1024);
    cvt_transpose_kernel<<<dim3(4096 / 32, 1024 / 32), 256, 0, stream>>>(w1,    w1T,   1024, 4096);
    cvt_transpose_kernel<<<dim3(1024 / 32, 4096 / 32), 256, 0, stream>>>(w2,    w2T,   4096, 1024);

    // LN1 -> QKV GEMM (bf16 out)
    ln_bf16_kernel<<<R, 256, 0, stream>>>(x, hln, 1024);
    gemm_bf16_kernel<0, 0, 0, 0><<<dim3(3072 / 128, R / 128), 256, 0, stream>>>(
        hln, wqkvT, 3072, 1024, nullptr, nullptr, nullptr, qkv);

    // attention
    attn_kernel<<<dim3(16, 16, 2), 256, 0, stream>>>(qkv, attn_o);

    // proj + residual(x) -> x1 fp32
    gemm_bf16_kernel<0, 0, 1, 1><<<dim3(1024 / 128, R / 128), 256, 0, stream>>>(
        attn_o, wfcT, 1024, 1024, nullptr, x, x1, nullptr);

    // LN2 -> MLP
    ln_bf16_kernel<<<R, 256, 0, stream>>>(x1, hln, 1024);
    gemm_bf16_kernel<1, 1, 0, 0><<<dim3(4096 / 128, R / 128), 256, 0, stream>>>(
        hln, w1T, 4096, 1024, b1, nullptr, nullptr, mlph);
    gemm_bf16_kernel<1, 0, 1, 1><<<dim3(1024 / 128, R / 128), 256, 0, stream>>>(
        mlph, w2T, 1024, 4096, b2, x1, out, nullptr);
}
